// GraphSAGE_55009941128032
// MI455X (gfx1250) — matrix operand order repaired and checked by reading, not compiled, not run on hardware
//
#include <hip/hip_runtime.h>

#define NNODES 100000
#define NEDGES 1600000
#define DIM 64
#define LDSW 68  // padded LDS row stride (floats) for transposed weights

typedef __attribute__((ext_vector_type(2)))  float  v2f;
typedef __attribute__((ext_vector_type(8)))  float  v8f;
typedef __attribute__((ext_vector_type(16))) __bf16 v16bf;

// ---------------------------------------------------------------- zero fill
__global__ void zero_f4(float4* __restrict__ p, int n4) {
  int i = blockIdx.x * blockDim.x + threadIdx.x;
  float4 z = make_float4(0.f, 0.f, 0.f, 0.f);
  for (; i < n4; i += (int)(gridDim.x * blockDim.x)) p[i] = z;
}

// ------------------------------------------------------- edge scatter (mean agg)
// One wave32 per edge: lane L handles channels L and L+32.
// Gather of x[src] row is two coalesced 128B reads; atomic adds to msg[dst]
// row are two coalesced 128B atomic bursts that resolve in L2 (msg fits in
// the 192MB L2 easily).
__global__ __launch_bounds__(256)
void sage_scatter(const float* __restrict__ x, const int* __restrict__ src,
                  const int* __restrict__ dst, float* __restrict__ msg,
                  float* __restrict__ deg, int count_deg) {
  unsigned tid  = blockIdx.x * blockDim.x + threadIdx.x;
  unsigned e    = tid >> 5;
  unsigned lane = tid & 31u;
  if (e >= NEDGES) return;
  int s = src[e];
  int d = dst[e];
  const float* xr = x + (size_t)s * DIM;
  float*       mr = msg + (size_t)d * DIM;
  float v0 = xr[lane];
  float v1 = xr[lane + 32];
  atomicAdd(mr + lane, v0);
  atomicAdd(mr + lane + 32, v1);
  if (count_deg && lane == 0) atomicAdd(deg + d, 1.0f);
}

// ------------------------------------------------------- bf16 fallback helpers
#if !__has_builtin(__builtin_amdgcn_wmma_f32_16x16x4_f32)
// A-matrix 16x32 bf16 layout (lane = 16*half + row):
//   VGPR v in 0..3 : K = 8*half + 2v + w       (+16 for v in 4..7)
__device__ inline v16bf a_frag_bf16(const float* __restrict__ row, int kb,
                                    int half, float scale) {
  v16bf a;
#pragma unroll
  for (int e = 0; e < 16; ++e) {
    int v = e >> 1, w = e & 1;
    int k = kb + ((v & 4) << 2) + 8 * half + 2 * (v & 3) + w;
    a[e] = (__bf16)(row[k] * scale);
  }
  return a;
}
// B-matrix 32x16 bf16 layout: element e -> K = 8*half + (e&7) (+16 for e >= 8).
__device__ inline v16bf b_frag_bf16(const float* __restrict__ wt_col, int kb,
                                    int half) {
  v16bf b;
#pragma unroll
  for (int e = 0; e < 16; ++e) b[e] = (__bf16)wt_col[kb + ((e & 8) << 1) + 8 * half + (e & 7)];
  return b;
}
#endif

// ------------------------------------------------------- fused SAGE layer GEMM
// out[n,:] = x[n,:] @ Wself + (msg[n,:]/max(deg[n],1)) @ Wneigh + bias
// Block = 8 waves = 32 rows x 64 cols of output; one 16x16 WMMA tile per wave.
// All A operands are register-preloaded BEFORE the barrier, so x==out
// (in-place) is safe.
__global__ __launch_bounds__(256)
void sage_gemm(const float* __restrict__ x, const float* __restrict__ msg,
               const float* __restrict__ deg, const float* __restrict__ Wself,
               const float* __restrict__ Wneigh, const float* __restrict__ bias,
               float* __restrict__ out, int has_neigh) {
  __shared__ float lw[2][DIM * LDSW];  // transposed weights: lw[m][c*LDSW + k]

  for (int idx = threadIdx.x; idx < DIM * DIM; idx += 256) {
    int r = idx >> 6, c = idx & 63;
    lw[0][c * LDSW + r] = Wself[idx];
    if (has_neigh) lw[1][c * LDSW + r] = Wneigh[idx];
  }

  int wave    = threadIdx.x >> 5;
  int lane    = threadIdx.x & 31;
  int half    = lane >> 4;
  int lrow    = lane & 15;
  int rowBase = blockIdx.x * 32 + (wave & 1) * 16;
  int colBase = (wave >> 1) * 16;
  int arow    = rowBase + lrow;
  int bcol    = colBase + lrow;
  const float* xrow = x + (size_t)arow * DIM;

  float rdeg = 1.0f;
  const float* mrow = nullptr;
  if (has_neigh) {
    mrow = msg + (size_t)arow * DIM;
    rdeg = 1.0f / fmaxf(deg[arow], 1.0f);
  }

#if __has_builtin(__builtin_amdgcn_wmma_f32_16x16x4_f32)
  // ---- exact-f32 path: V_WMMA_F32_16X16X4_F32, 16 K-steps per matmul ----
  float as[32], an[32];
#pragma unroll
  for (int i = 0; i < 16; ++i) {
    int k0 = 4 * i + 2 * half;
    v2f t = *(const v2f*)(xrow + k0);
    as[2 * i] = t.x;
    as[2 * i + 1] = t.y;
    if (has_neigh) {
      v2f m = *(const v2f*)(mrow + k0);
      an[2 * i] = m.x * rdeg;
      an[2 * i + 1] = m.y * rdeg;
    }
  }
  __syncthreads();

  v8f c = {};
  const float* bs = &lw[0][bcol * LDSW];
#pragma unroll
  for (int i = 0; i < 16; ++i) {
    int k0 = 4 * i + 2 * half;
    v2f a;
    a.x = as[2 * i];
    a.y = as[2 * i + 1];
    v2f b = *(const v2f*)(bs + k0);  // ds_load_b64, conflict-free (stride 68)
    c = __builtin_amdgcn_wmma_f32_16x16x4_f32(false, a, false, b, (short)0, c,
                                              false, false);
  }
  if (has_neigh) {
    const float* bn = &lw[1][bcol * LDSW];
#pragma unroll
    for (int i = 0; i < 16; ++i) {
      int k0 = 4 * i + 2 * half;
      v2f a;
      a.x = an[2 * i];
      a.y = an[2 * i + 1];
      v2f b = *(const v2f*)(bn + k0);
      c = __builtin_amdgcn_wmma_f32_16x16x4_f32(false, a, false, b, (short)0, c,
                                                false, false);
    }
  }
#else
  // ---- fallback: bf16 inputs, f32 accumulation (V_WMMA_F32_16X16X32_BF16) ----
  v16bf a0 = a_frag_bf16(xrow, 0, half, 1.0f);
  v16bf a1 = a_frag_bf16(xrow, 32, half, 1.0f);
  v16bf n0{}, n1{};
  if (has_neigh) {
    n0 = a_frag_bf16(mrow, 0, half, rdeg);
    n1 = a_frag_bf16(mrow, 32, half, rdeg);
  }
  __syncthreads();

  v8f c = {};
  const float* bs = &lw[0][bcol * LDSW];
  c = __builtin_amdgcn_wmma_f32_16x16x32_bf16(false, a0, false,
                                              b_frag_bf16(bs, 0, half),
                                              (short)0, c, false, false);
  c = __builtin_amdgcn_wmma_f32_16x16x32_bf16(false, a1, false,
                                              b_frag_bf16(bs, 32, half),
                                              (short)0, c, false, false);
  if (has_neigh) {
    const float* bn = &lw[1][bcol * LDSW];
    c = __builtin_amdgcn_wmma_f32_16x16x32_bf16(false, n0, false,
                                                b_frag_bf16(bn, 0, half),
                                                (short)0, c, false, false);
    c = __builtin_amdgcn_wmma_f32_16x16x32_bf16(false, n1, false,
                                                b_frag_bf16(bn, 32, half),
                                                (short)0, c, false, false);
  }
#endif

  // C/D layout: VGPR r, lane l -> row = r + 8*(l>>4), col = l&15
  float bv = bias[colBase + lrow];
#pragma unroll
  for (int r = 0; r < 8; ++r) {
    int row = rowBase + r + 8 * half;
    out[(size_t)row * DIM + colBase + lrow] = c[r] + bv;
  }
}

// ---------------------------------------------------------------- launcher
extern "C" void kernel_launch(void* const* d_in, const int* in_sizes, int n_in,
                              void* d_out, int out_size, void* d_ws,
                              size_t ws_size, hipStream_t stream) {
  (void)in_sizes; (void)n_in; (void)out_size; (void)ws_size;
  const float* feat = (const float*)d_in[0];
  const int*   esrc = (const int*)d_in[1];
  const int*   edst = (const int*)d_in[2];
  const float* Ws0  = (const float*)d_in[3];
  const float* Wn0  = (const float*)d_in[4];
  const float* b0   = (const float*)d_in[5];
  const float* Ws1  = (const float*)d_in[6];
  const float* Wn1  = (const float*)d_in[7];
  const float* b1   = (const float*)d_in[8];
  const float* Wfc  = (const float*)d_in[9];
  const float* bfc  = (const float*)d_in[10];
  float* out = (float*)d_out;

  // workspace: [msg: N*64][deg: N][h: N*64]  (~52 MB)
  float* msg = (float*)d_ws;
  float* deg = msg + (size_t)NNODES * DIM;
  float* h   = deg + NNODES;

  const int scatterBlocks = (NEDGES * 32) / 256;  // 200000
  const int gemmBlocks    = NNODES / 32;          // 3125

  // ---- layer 0 ----
  {
    int n4 = (NNODES * DIM + NNODES) / 4;  // msg + deg contiguous
    zero_f4<<<(n4 + 255) / 256, 256, 0, stream>>>((float4*)msg, n4);
  }
  sage_scatter<<<scatterBlocks, 256, 0, stream>>>(feat, esrc, edst, msg, deg, 1);
  sage_gemm<<<gemmBlocks, 256, 0, stream>>>(feat, msg, deg, Ws0, Wn0, b0, h, 1);

  // ---- layer 1 (deg reused; GEMM runs in-place h -> h) ----
  {
    int n4 = (NNODES * DIM) / 4;
    zero_f4<<<(n4 + 255) / 256, 256, 0, stream>>>((float4*)msg, n4);
  }
  sage_scatter<<<scatterBlocks, 256, 0, stream>>>(h, esrc, edst, msg, deg, 0);
  sage_gemm<<<gemmBlocks, 256, 0, stream>>>(h, msg, deg, Ws1, Wn1, b1, h, 1);

  // ---- final FC: out = h @ Wfc + bfc ----
  sage_gemm<<<gemmBlocks, 256, 0, stream>>>(h, nullptr, nullptr, Wfc, nullptr,
                                            bfc, out, 0);
}
